// MFC_RMF_1683627180567
// MI455X (gfx1250) — matrix operand order repaired and checked by reading, not compiled, not run on hardware
//
#include <hip/hip_runtime.h>
#include <math.h>

// ---------------- problem sizes ----------------
#define NV 8192
#define FV 128
#define HV 512
#define DV 256
#define MSV 256
#define EV 131072
#define TEV (2 * EV + NV) /* 270336 symmetrized edges + self loops */
#define EPSF 1e-8f

typedef _Float16 half_t;
typedef _Float16 v16h __attribute__((ext_vector_type(16)));
typedef _Float16 v8h  __attribute__((ext_vector_type(8)));
typedef float    v8f  __attribute__((ext_vector_type(8)));

__device__ __forceinline__ v8f zero8() {
  v8f z;
#pragma unroll
  for (int i = 0; i < 8; i++) z[i] = 0.0f;
  return z;
}

__device__ __forceinline__ v8f wmma_f16(v16h a, v16h b, v8f c) {
#if defined(__gfx1250__)
  // 8 args: (neg_a, A, neg_b, B, c_mod, C, reuse_a, reuse_b)
  return __builtin_amdgcn_wmma_f32_16x16x32_f16(false, a, false, b, (short)0, c,
                                                false, false);
#else
  (void)a; (void)b;
  return c;
#endif
}

// A fragment: 16x32 f16 (MxK), A row-major [*, lda] half.
// lane l: M = l&15, hi = l>>4; halfs 0..7 <- K = k0+8*hi .. +7 ; halfs 8..15 <- K = k0+16+8*hi .. +7
__device__ __forceinline__ v16h load_a_rm(const half_t* A, int lda, int m0, int k0) {
  int l = threadIdx.x & 31;
  int m = l & 15, hi = l >> 4;
  const half_t* r = A + (size_t)(m0 + m) * lda + k0;
  v8h lo = *(const v8h*)(r + 8 * hi);
  v8h hh = *(const v8h*)(r + 16 + 8 * hi);
  v16h o;
#pragma unroll
  for (int i = 0; i < 8; i++) { o[i] = lo[i]; o[8 + i] = hh[i]; }
  return o;
}

// B fragment: 32x16 f16 (KxN), with B[k][n] == Bt[n][k], Bt row-major [*, ldbt] half.
// lane l: N = n0 + (l&15); halfs 0..7 <- K = k0+8*hi .. +7 ; halfs 8..15 <- K = k0+16+8*hi .. +7
__device__ __forceinline__ v16h load_b_tn(const half_t* Bt, int ldbt, int k0, int n0) {
  int l = threadIdx.x & 31;
  int n = l & 15, hi = l >> 4;
  const half_t* r = Bt + (size_t)(n0 + n) * ldbt + k0;
  v8h lo = *(const v8h*)(r + 8 * hi);
  v8h hh = *(const v8h*)(r + 16 + 8 * hi);
  v16h o;
#pragma unroll
  for (int i = 0; i < 8; i++) { o[i] = lo[i]; o[8 + i] = hh[i]; }
  return o;
}

// C/D fragment store: lane l holds col n0+(l&15), rows m0 + v + 8*(l>>4) in VGPR v.
__device__ __forceinline__ void store_c(float* C, int ldc, int m0, int n0, v8f acc) {
  int l = threadIdx.x & 31;
  int hi = l >> 4;
  int col = n0 + (l & 15);
#pragma unroll
  for (int v = 0; v < 8; v++) C[(size_t)(m0 + v + 8 * hi) * ldc + col] = acc[v];
}

// ---------------- WMMA GEMM: C[M,Nc] f32 = A[M,K] half @ Bt[Nc,K]^T half ----------------
// block = 128 threads (4 waves); wave computes 16(M) x 64(N); block tile 64x64.
__global__ __launch_bounds__(128) void k_gemm(const half_t* __restrict__ A,
                                              const half_t* __restrict__ Bt,
                                              float* __restrict__ C,
                                              int M, int Nc, int K) {
  int wave = threadIdx.x >> 5;
  int m0 = blockIdx.y * 64 + wave * 16;
  int n0 = blockIdx.x * 64;
  v8f a0 = zero8(), a1 = zero8(), a2 = zero8(), a3 = zero8();
  for (int k0 = 0; k0 < K; k0 += 32) {
    v16h av = load_a_rm(A, K, m0, k0);
    a0 = wmma_f16(av, load_b_tn(Bt, K, k0, n0), a0);
    a1 = wmma_f16(av, load_b_tn(Bt, K, k0, n0 + 16), a1);
    a2 = wmma_f16(av, load_b_tn(Bt, K, k0, n0 + 32), a2);
    a3 = wmma_f16(av, load_b_tn(Bt, K, k0, n0 + 48), a3);
  }
  store_c(C, Nc, m0, n0, a0);
  store_c(C, Nc, m0, n0 + 16, a1);
  store_c(C, Nc, m0, n0 + 32, a2);
  store_c(C, Nc, m0, n0 + 48, a3);
}

// ---------------- mutual pass 1: colsum[m] = sum_n cos[n,m] ----------------
__global__ __launch_bounds__(128) void k_cos_colsum(const half_t* __restrict__ Ah,
                                                    const half_t* __restrict__ Bh,
                                                    const float* __restrict__ na,
                                                    const float* __restrict__ nb,
                                                    float* __restrict__ colsum,
                                                    int N, int D) {
  int wave = threadIdx.x >> 5;
  int n0 = blockIdx.y * 64 + wave * 16;
  int m0 = blockIdx.x * 64;
  v8f acc[4];
#pragma unroll
  for (int j = 0; j < 4; j++) acc[j] = zero8();
  for (int k0 = 0; k0 < D; k0 += 32) {
    v16h av = load_a_rm(Ah, D, n0, k0);
#pragma unroll
    for (int j = 0; j < 4; j++)
      acc[j] = wmma_f16(av, load_b_tn(Bh, D, k0, m0 + 16 * j), acc[j]);
  }
  int l = threadIdx.x & 31, hi = l >> 4, lo = l & 15;
  float nr[8];
#pragma unroll
  for (int v = 0; v < 8; v++) nr[v] = na[n0 + v + 8 * hi];
#pragma unroll
  for (int j = 0; j < 4; j++) {
    int col = m0 + 16 * j + lo;
    float nc = nb[col];
    float s = 0.0f;
#pragma unroll
    for (int v = 0; v < 8; v++) s += acc[j][v] / fmaxf(nr[v] * nc, EPSF);
    s += __shfl_xor(s, 16);
    if (hi == 0) atomicAdd(&colsum[col], s);
  }
}

// ---------------- mutual pass 2 (fused, flash-style): Out = (cos/colsum) @ h2 ----------------
// Ah = h1 half [N,D]; Bh = h2 half [N,D]; Bt2 = h2^T half [D,N].
// Wave owns 16 output rows x 256 output cols (16 f32 accum frags); loops m in steps of 32:
// cos tile (2 WMMA frags over full D), scale to coe, round-trip through per-wave LDS to
// reshape into an A fragment (f16), then 16 WMMAs against h2 rows m..m+31.
__global__ __launch_bounds__(128) void k_mutual_out(const half_t* __restrict__ Ah,
                                                    const half_t* __restrict__ Bh,
                                                    const half_t* __restrict__ Bt2,
                                                    const float* __restrict__ na,
                                                    const float* __restrict__ nb,
                                                    const float* __restrict__ colsum,
                                                    float* __restrict__ Out,
                                                    int N, int D) {
  __shared__ __align__(16) half_t smem[4][16 * 32];
  int wave = threadIdx.x >> 5;
  int n0 = blockIdx.x * 64 + wave * 16;
  int l = threadIdx.x & 31, hi = l >> 4, lo = l & 15;
  half_t* tile = &smem[wave][0];

  v8f out[16];
#pragma unroll
  for (int t = 0; t < 16; t++) out[t] = zero8();

  float nr[8];
#pragma unroll
  for (int v = 0; v < 8; v++) nr[v] = na[n0 + v + 8 * hi];

  for (int m0 = 0; m0 < N; m0 += 32) {
    v8f c0 = zero8(), c1 = zero8();
    for (int k0 = 0; k0 < D; k0 += 32) {
      v16h av = load_a_rm(Ah, D, n0, k0);
      c0 = wmma_f16(av, load_b_tn(Bh, D, k0, m0), c0);
      c1 = wmma_f16(av, load_b_tn(Bh, D, k0, m0 + 16), c1);
    }
    int col0 = m0 + lo, col1 = m0 + 16 + lo;
    float i0 = 1.0f / colsum[col0], i1 = 1.0f / colsum[col1];
    float b0 = nb[col0], b1 = nb[col1];
#pragma unroll
    for (int v = 0; v < 8; v++) {
      int row = v + 8 * hi;
      float w0 = (c0[v] / fmaxf(nr[v] * b0, EPSF)) * i0;
      float w1 = (c1[v] / fmaxf(nr[v] * b1, EPSF)) * i1;
      tile[row * 32 + lo] = (half_t)w0;
      tile[row * 32 + 16 + lo] = (half_t)w1;
    }
    // LDS ops from one wave are in-order (DScnt); safe to read back within the wave.
    v16h ca = load_a_rm(tile, 32, 0, 0);
#pragma unroll
    for (int t = 0; t < 16; t++)
      out[t] = wmma_f16(ca, load_b_tn(Bt2, N, m0, 16 * t), out[t]);
  }
#pragma unroll
  for (int t = 0; t < 16; t++) store_c(Out, D, n0, 16 * t, out[t]);
}

// ---------------- small utility kernels ----------------
__global__ void k_fill(float* p, float v, int n) {
  int i = blockIdx.x * blockDim.x + threadIdx.x;
  if (i < n) p[i] = v;
}

__global__ void k_f2h(const float* a, half_t* o, int n) {
  int i = blockIdx.x * blockDim.x + threadIdx.x;
  if (i < n) o[i] = (half_t)a[i];
}

// transpose+convert: src f32 [R,C] -> dst half [C,R]; optional square
__global__ void k_transpose_f2h(const float* __restrict__ src, half_t* __restrict__ dst,
                                int R, int C, int sq) {
  __shared__ float t[32][33];
  int c0 = blockIdx.x * 32, r0 = blockIdx.y * 32;
  int tx = threadIdx.x, ty = threadIdx.y;  // (32,8)
#pragma unroll
  for (int i = 0; i < 32; i += 8) {
    float v = src[(size_t)(r0 + ty + i) * C + c0 + tx];
    t[ty + i][tx] = sq ? v * v : v;
  }
  __syncthreads();
#pragma unroll
  for (int i = 0; i < 32; i += 8)
    dst[(size_t)(c0 + ty + i) * R + r0 + tx] = (half_t)t[tx][ty + i];
}

__global__ void k_stats(const float* __restrict__ x, int n, float* st) {
  __shared__ float s0[256], s1[256];
  float a = 0.0f, b = 0.0f;
  for (int i = blockIdx.x * blockDim.x + threadIdx.x; i < n; i += gridDim.x * blockDim.x) {
    float v = x[i];
    a += v; b += v * v;
  }
  s0[threadIdx.x] = a; s1[threadIdx.x] = b;
  __syncthreads();
  for (int o = 128; o; o >>= 1) {
    if (threadIdx.x < o) { s0[threadIdx.x] += s0[threadIdx.x + o]; s1[threadIdx.x] += s1[threadIdx.x + o]; }
    __syncthreads();
  }
  if (!threadIdx.x) { atomicAdd(&st[0], s0[0]); atomicAdd(&st[1], s1[0]); }
}

__global__ void k_norm2h(const float* __restrict__ x, const float* st, int n, half_t* o) {
  int i = blockIdx.x * blockDim.x + threadIdx.x;
  if (i >= n) return;
  float nn = (float)n;
  float mean = st[0] / nn;
  float var = (st[1] - st[0] * st[0] / nn) / (nn - 1.0f);  // ddof=1
  float inv = rsqrtf(var);
  o[i] = (half_t)((x[i] - mean) * inv);
}

__global__ void k_elu2h(const float* __restrict__ x, half_t* o, int n) {
  int i = blockIdx.x * blockDim.x + threadIdx.x;
  if (i >= n) return;
  float v = x[i];
  o[i] = (half_t)(v > 0.0f ? v : expf(v) - 1.0f);
}

// mode 0: o = a*b elementwise ; mode 1: b is a [C] row vector broadcast over rows
__global__ void k_prod2h(const float* __restrict__ a, const float* __restrict__ b,
                         half_t* o, int n, int C, int mode) {
  int i = blockIdx.x * blockDim.x + threadIdx.x;
  if (i >= n) return;
  float bb = (mode == 1) ? b[i % C] : b[i];
  o[i] = (half_t)(a[i] * bb);
}

__global__ void k_combine(float* dot, const float* __restrict__ n1, int b1,
                          const float* __restrict__ n2, int b2, int n, int C) {
  int i = blockIdx.x * blockDim.x + threadIdx.x;
  if (i >= n) return;
  int c = i % C;
  float x1 = n1[b1 ? c : i], x2 = n2[b2 ? c : i];
  dot[i] = dot[i] / fmaxf(sqrtf(x1) * sqrtf(x2), EPSF);
}

// per-row dual dot products: oa[r]=h[r,:].va  ob[r]=h[r,:].vb  (wave per row)
__global__ void k_rowdots(const float* __restrict__ h, const float* __restrict__ va,
                          const float* __restrict__ vb, float* oa, float* ob,
                          int N, int Dm) {
  int wave = threadIdx.x >> 5, lane = threadIdx.x & 31;
  int row = blockIdx.x * 8 + wave;
  if (row >= N) return;
  const float* r = h + (size_t)row * Dm;
  float s1 = 0.0f, s2 = 0.0f;
  for (int c = lane; c < Dm; c += 32) { float x = r[c]; s1 += x * va[c]; s2 += x * vb[c]; }
  for (int o = 16; o; o >>= 1) { s1 += __shfl_xor(s1, o); s2 += __shfl_xor(s2, o); }
  if (!lane) { oa[row] = s1; ob[row] = s2; }
}

__global__ void k_rownorm(const float* __restrict__ h, float* nv, int N, int Dm) {
  int wave = threadIdx.x >> 5, lane = threadIdx.x & 31;
  int row = blockIdx.x * 8 + wave;
  if (row >= N) return;
  const float* r = h + (size_t)row * Dm;
  float s = 0.0f;
  for (int c = lane; c < Dm; c += 32) { float x = r[c]; s += x * x; }
  for (int o = 16; o; o >>= 1) s += __shfl_xor(s, o);
  if (!lane) nv[row] = sqrtf(s);
}

// ---------------- GAT edge softmax + aggregation ----------------
__device__ __forceinline__ void edge_sd(int e, const int* __restrict__ ei, int& s, int& d) {
  if (e < EV)            { s = ei[e];            d = ei[EV + e]; }
  else if (e < 2 * EV)   { s = ei[EV + (e - EV)]; d = ei[e - EV]; }
  else                   { s = d = e - 2 * EV; }
}

__device__ __forceinline__ float leaky02(float x) { return x > 0.0f ? x : 0.2f * x; }

__device__ __forceinline__ unsigned fmap(float f) {
  unsigned u = __float_as_uint(f);
  return (u & 0x80000000u) ? ~u : (u | 0x80000000u);
}
__device__ __forceinline__ float funmap(unsigned u) {
  return __uint_as_float((u & 0x80000000u) ? (u & 0x7fffffffu) : ~u);
}

__global__ void k_edge_max(const int* __restrict__ ei, const float* __restrict__ asrc,
                           const float* __restrict__ adst, unsigned* mmax) {
  int e = blockIdx.x * blockDim.x + threadIdx.x;
  if (e >= TEV) return;
  int s, d; edge_sd(e, ei, s, d);
  atomicMax(&mmax[d], fmap(leaky02(asrc[s] + adst[d])));
}

__global__ void k_edge_expsum(const int* __restrict__ ei, const float* __restrict__ asrc,
                              const float* __restrict__ adst, const unsigned* __restrict__ mmax,
                              float* den, float* exv) {
  int e = blockIdx.x * blockDim.x + threadIdx.x;
  if (e >= TEV) return;
  int s, d; edge_sd(e, ei, s, d);
  float ex = expf(leaky02(asrc[s] + adst[d]) - funmap(mmax[d]));
  exv[e] = ex;
  atomicAdd(&den[d], ex);
}

__global__ void k_edge_aggr(const int* __restrict__ ei, const float* __restrict__ exv,
                            const float* __restrict__ den, const float* __restrict__ h,
                            float* out, int Dm) {
  int e = blockIdx.x;
  int s, d; edge_sd(e, ei, s, d);
  float alpha = exv[e] / (den[d] + 1e-16f);
  const float* hr = h + (size_t)s * Dm;
  float* orow = out + (size_t)d * Dm;
  for (int c = threadIdx.x; c < Dm; c += blockDim.x)
    atomicAdd(&orow[c], alpha * hr[c]);
}

// ---------------- readout pieces ----------------
__global__ void k_colacc(const float* __restrict__ h, float* colsum, int N, int C) {
  int c = threadIdx.x;                    // C == 256
  int r0 = blockIdx.x * 128;
  float s = 0.0f;
  for (int r = 0; r < 128; r++) s += h[(size_t)(r0 + r) * C + c];
  atomicAdd(&colsum[c], s);
}

__global__ void k_gvec(const float* __restrict__ cm, const float* __restrict__ Wr,
                       float invN, float* g, int C) {
  int m = threadIdx.x;
  float s = 0.0f;
  for (int d = 0; d < C; d++) s += cm[d] * invN * Wr[d * C + m];
  g[m] = tanhf(s);
}

__global__ void k_wsum(const float* __restrict__ h, const float* __restrict__ g,
                       float* outv, int N, int C) {
  __shared__ float acc[256];
  acc[threadIdx.x] = 0.0f;
  __syncthreads();
  int wave = threadIdx.x >> 5, lane = threadIdx.x & 31;
  for (int rr = 0; rr < 4; rr++) {
    int row = blockIdx.x * 32 + wave * 4 + rr;
    const float* r = h + (size_t)row * C;
    float s = 0.0f;
    for (int c = lane; c < C; c += 32) s += r[c] * g[c];
    for (int o = 16; o; o >>= 1) s += __shfl_xor(s, o);
    float sv = 1.0f / (1.0f + expf(-s));
    for (int c = lane; c < C; c += 32) atomicAdd(&acc[c], sv * r[c]);
  }
  __syncthreads();
  atomicAdd(&outv[threadIdx.x], acc[threadIdx.x]);
}

// ---------------- small match helpers ----------------
// out[m] = sum_d v[d]^2 * W[d,m]^2
__global__ void k_sqvecmat(const float* __restrict__ v, const float* __restrict__ W,
                           float* o, int C) {
  int m = threadIdx.x;
  float s = 0.0f;
  for (int d = 0; d < C; d++) { float w = W[d * C + m]; s += v[d] * v[d] * w * w; }
  o[m] = s;
}

// full _match for two [1,C] vectors
__global__ void k_om(const float* __restrict__ a, const float* __restrict__ b,
                     const float* __restrict__ W, float* o, int C) {
  int m = threadIdx.x;
  float sd = 0.0f, s1 = 0.0f, s2 = 0.0f;
  for (int d = 0; d < C; d++) {
    float w = W[d * C + m]; float w2 = w * w;
    sd += a[d] * b[d] * w2; s1 += a[d] * a[d] * w2; s2 += b[d] * b[d] * w2;
  }
  o[m] = sd / fmaxf(sqrtf(s1) * sqrtf(s2), EPSF);
}

// act: 0=none 1=relu 2=sigmoid
__global__ void k_dense(const float* __restrict__ in, const float* __restrict__ W,
                        const float* __restrict__ b, float* out, int IN, int OUT, int act) {
  int o = blockIdx.x * blockDim.x + threadIdx.x;
  if (o >= OUT) return;
  float s = b[o];
  for (int i = 0; i < IN; i++) s += in[i] * W[i * OUT + o];
  if (act == 1) s = fmaxf(s, 0.0f);
  else if (act == 2) s = 1.0f / (1.0f + expf(-s));
  out[o] = s;
}

__global__ void k_final(const float* __restrict__ z, const int* __restrict__ label,
                        float* out, int out_size) {
  if (threadIdx.x || blockIdx.x) return;
  out[0] = z[0];
  if (out_size > 1) out[1] = (float)label[0];
  if (out_size > 2) out[2] = expf(-(float)label[0]);
}

// ================================================================
extern "C" void kernel_launch(void* const* d_in, const int* in_sizes, int n_in,
                              void* d_out, int out_size, void* d_ws, size_t ws_size,
                              hipStream_t stream) {
  (void)in_sizes; (void)n_in; (void)ws_size;
  const float* x_s    = (const float*)d_in[0];
  const float* x_t    = (const float*)d_in[1];
  const int*   ei_s   = (const int*)d_in[2];
  const int*   ei_t   = (const int*)d_in[3];
  const int*   label  = (const int*)d_in[4];
  const float* W_g1   = (const float*)d_in[5];
  const float* a1s    = (const float*)d_in[6];
  const float* a1d    = (const float*)d_in[7];
  const float* W_g2   = (const float*)d_in[8];
  const float* a2s    = (const float*)d_in[9];
  const float* a2d    = (const float*)d_in[10];
  const float* W_read = (const float*)d_in[11];
  const float* W_match= (const float*)d_in[12];
  const float* mw1 = (const float*)d_in[13]; const float* mb1 = (const float*)d_in[14];
  const float* mw2 = (const float*)d_in[15]; const float* mb2 = (const float*)d_in[16];
  const float* mw3 = (const float*)d_in[17]; const float* mb3 = (const float*)d_in[18];
  const float* mw4 = (const float*)d_in[19]; const float* mb4 = (const float*)d_in[20];

  // ---- workspace carve-up ----
  char* wp = (char*)d_ws;
  auto alloc = [&](size_t bytes) -> void* {
    void* p = (void*)wp;
    wp += (bytes + 255) & ~(size_t)255;
    return p;
  };
  float*  h1f   = (float*)alloc((size_t)NV * DV * 4);
  float*  h2f   = (float*)alloc((size_t)NV * DV * 4);
  float*  h1m   = (float*)alloc((size_t)NV * DV * 4);
  float*  h2m   = (float*)alloc((size_t)NV * DV * 4);
  half_t* h1h   = (half_t*)alloc((size_t)NV * DV * 2);
  half_t* h2h   = (half_t*)alloc((size_t)NV * DV * 2);
  half_t* h1t   = (half_t*)alloc((size_t)NV * DV * 2);
  half_t* h2t   = (half_t*)alloc((size_t)NV * DV * 2);
  half_t* bfA   = (half_t*)alloc((size_t)NV * HV * 2);
  half_t* w1t   = (half_t*)alloc((size_t)FV * HV * 2);   // [HV,FV]
  half_t* w2t   = (half_t*)alloc((size_t)HV * DV * 2);   // [DV,HV]
  half_t* wmt   = (half_t*)alloc((size_t)DV * MSV * 2);  // (W_match^2)^T [MSV,DV]
  float*  tmpA  = (float*)alloc((size_t)NV * HV * 4);
  float*  tmpB  = (float*)alloc((size_t)NV * HV * 4);
  float*  tmpC  = (float*)alloc((size_t)NV * DV * 4);
  float*  tmpD  = (float*)alloc((size_t)NV * DV * 4);
  float*  asrc  = (float*)alloc(NV * 4);
  float*  adst  = (float*)alloc(NV * 4);
  float*  mmaxb = (float*)alloc(NV * 4);
  float*  den   = (float*)alloc(NV * 4);
  float*  exv   = (float*)alloc((size_t)TEV * 4);
  float*  n1v   = (float*)alloc(NV * 4);
  float*  n2v   = (float*)alloc(NV * 4);
  float*  colsum= (float*)alloc(NV * 4);
  float*  stats = (float*)alloc(8 * 4);
  float*  cmean = (float*)alloc(DV * 4);
  float*  gvec  = (float*)alloc(DV * 4);
  float*  nrow  = (float*)alloc(DV * 4);
  float*  gslot = (float*)alloc(4 * DV * 4);   // h1g, h2g, h1mg, h2mg
  float*  zbuf  = (float*)alloc(8 * MSV * 4);  // MLP input, 2048
  float*  mlp1  = (float*)alloc(512 * 4);
  float*  mlp2  = (float*)alloc(256 * 4);
  float*  mlp3  = (float*)alloc(128 * 4);
  float*  mlp4  = (float*)alloc(4);

  auto fill = [&](float* p, float v, int n) {
    k_fill<<<(n + 255) / 256, 256, 0, stream>>>(p, v, n);
  };
  auto gemm = [&](const half_t* A, const half_t* Bt, float* C, int M, int Nc, int K) {
    k_gemm<<<dim3(Nc / 64, M / 64), 128, 0, stream>>>(A, Bt, C, M, Nc, K);
  };
  auto readout = [&](const float* src, float* outvec) {
    fill(cmean, 0.0f, DV);
    k_colacc<<<NV / 128, DV, 0, stream>>>(src, cmean, NV, DV);
    k_gvec<<<1, DV, 0, stream>>>(cmean, W_read, 1.0f / NV, gvec, DV);
    fill(outvec, 0.0f, DV);
    k_wsum<<<NV / 32, 256, 0, stream>>>(src, gvec, outvec, NV, DV);
  };

  // ---- one-time weight prep (half + transpose; W_match squared) ----
  k_transpose_f2h<<<dim3(HV / 32, FV / 32), dim3(32, 8), 0, stream>>>(W_g1, w1t, FV, HV, 0);
  k_transpose_f2h<<<dim3(DV / 32, HV / 32), dim3(32, 8), 0, stream>>>(W_g2, w2t, HV, DV, 0);
  k_transpose_f2h<<<dim3(MSV / 32, DV / 32), dim3(32, 8), 0, stream>>>(W_match, wmt, DV, MSV, 1);

  // ---- 2-layer GAT per graph ----
  auto gat = [&](const float* x, const int* ei, float* hout) {
    fill(stats, 0.0f, 2);
    k_stats<<<256, 256, 0, stream>>>(x, NV * FV, stats);
    k_norm2h<<<(NV * FV + 255) / 256, 256, 0, stream>>>(x, stats, NV * FV, bfA);
    gemm(bfA, w1t, tmpA, NV, HV, FV);                                   // h = f @ W1
    k_rowdots<<<NV / 8, 256, 0, stream>>>(tmpA, a1s, a1d, asrc, adst, NV, HV);
    fill(mmaxb, 0.0f, NV); fill(den, 0.0f, NV);
    k_edge_max<<<(TEV + 255) / 256, 256, 0, stream>>>(ei, asrc, adst, (unsigned*)mmaxb);
    k_edge_expsum<<<(TEV + 255) / 256, 256, 0, stream>>>(ei, asrc, adst,
                                                         (const unsigned*)mmaxb, den, exv);
    fill(tmpB, 0.0f, NV * HV);
    k_edge_aggr<<<TEV, 256, 0, stream>>>(ei, exv, den, tmpA, tmpB, HV);
    k_elu2h<<<(NV * HV + 255) / 256, 256, 0, stream>>>(tmpB, bfA, NV * HV);
    gemm(bfA, w2t, tmpA, NV, DV, HV);                                   // h2 = elu(h) @ W2
    k_rowdots<<<NV / 8, 256, 0, stream>>>(tmpA, a2s, a2d, asrc, adst, NV, DV);
    fill(mmaxb, 0.0f, NV); fill(den, 0.0f, NV);
    k_edge_max<<<(TEV + 255) / 256, 256, 0, stream>>>(ei, asrc, adst, (unsigned*)mmaxb);
    k_edge_expsum<<<(TEV + 255) / 256, 256, 0, stream>>>(ei, asrc, adst,
                                                         (const unsigned*)mmaxb, den, exv);
    fill(hout, 0.0f, NV * DV);
    k_edge_aggr<<<TEV, 256, 0, stream>>>(ei, exv, den, tmpA, hout, DV);
  };
  gat(x_s, ei_s, h1f);
  gat(x_t, ei_t, h2f);

  // ---- half copies / transposes / norms for mutual attention ----
  k_f2h<<<(NV * DV + 255) / 256, 256, 0, stream>>>(h1f, h1h, NV * DV);
  k_f2h<<<(NV * DV + 255) / 256, 256, 0, stream>>>(h2f, h2h, NV * DV);
  k_transpose_f2h<<<dim3(DV / 32, NV / 32), dim3(32, 8), 0, stream>>>(h1f, h1t, NV, DV, 0);
  k_transpose_f2h<<<dim3(DV / 32, NV / 32), dim3(32, 8), 0, stream>>>(h2f, h2t, NV, DV, 0);
  k_rownorm<<<NV / 8, 256, 0, stream>>>(h1f, n1v, NV, DV);
  k_rownorm<<<NV / 8, 256, 0, stream>>>(h2f, n2v, NV, DV);

  // ---- mutual: h1m = coe(h1,h2) @ h2 ; h2m = coe(h2,h1) @ h1 ----
  fill(colsum, 0.0f, NV);
  k_cos_colsum<<<dim3(NV / 64, NV / 64), 128, 0, stream>>>(h1h, h2h, n1v, n2v, colsum, NV, DV);
  k_mutual_out<<<NV / 64, 128, 0, stream>>>(h1h, h2h, h2t, n1v, n2v, colsum, h1m, NV, DV);
  fill(colsum, 0.0f, NV);
  k_cos_colsum<<<dim3(NV / 64, NV / 64), 128, 0, stream>>>(h2h, h1h, n2v, n1v, colsum, NV, DV);
  k_mutual_out<<<NV / 64, 128, 0, stream>>>(h2h, h1h, h1t, n2v, n1v, colsum, h2m, NV, DV);

  // ---- graph-level readouts ----
  float* g_h1  = gslot + 0 * DV;
  float* g_h2  = gslot + 1 * DV;
  float* g_h1m = gslot + 2 * DV;
  float* g_h2m = gslot + 3 * DV;
  readout(h1f, g_h1);
  readout(h2f, g_h2);
  readout(h1m, g_h1m);
  readout(h2m, g_h2m);

  // ---- matching variants (miu / phi / psi / om), each readout -> zbuf slot ----
  int nND = NV * DV;
  auto matchNN = [&](const float* a, const float* b, float* slot) {
    k_prod2h<<<(nND + 255) / 256, 256, 0, stream>>>(a, b, bfA, nND, DV, 0);
    gemm(bfA, wmt, tmpB, NV, MSV, DV);
    k_prod2h<<<(nND + 255) / 256, 256, 0, stream>>>(a, a, bfA, nND, DV, 0);
    gemm(bfA, wmt, tmpC, NV, MSV, DV);
    k_prod2h<<<(nND + 255) / 256, 256, 0, stream>>>(b, b, bfA, nND, DV, 0);
    gemm(bfA, wmt, tmpD, NV, MSV, DV);
    k_combine<<<(nND + 255) / 256, 256, 0, stream>>>(tmpB, tmpC, 0, tmpD, 0, nND, MSV);
    readout(tmpB, slot);
  };
  auto matchVec = [&](const float* mat, const float* vec, float* slot) {
    // _match(mat[N,D], vec[1,D]) (and symmetric case: product & norms commute)
    k_prod2h<<<(nND + 255) / 256, 256, 0, stream>>>(mat, vec, bfA, nND, DV, 1);
    gemm(bfA, wmt, tmpB, NV, MSV, DV);
    k_prod2h<<<(nND + 255) / 256, 256, 0, stream>>>(mat, mat, bfA, nND, DV, 0);
    gemm(bfA, wmt, tmpC, NV, MSV, DV);
    k_sqvecmat<<<1, MSV, 0, stream>>>(vec, W_match, nrow, DV);
    k_combine<<<(nND + 255) / 256, 256, 0, stream>>>(tmpB, tmpC, 0, nrow, 1, nND, MSV);
    readout(tmpB, slot);
  };
  matchNN(h1f, h1m, zbuf + 0 * MSV);            // miu1g
  matchVec(h1f, g_h1m, zbuf + 1 * MSV);         // phi1g  (_match(h1, h1mg))
  matchVec(h1m, g_h1, zbuf + 2 * MSV);          // psi1g  (_match(h1g, h1m))
  k_om<<<1, MSV, 0, stream>>>(g_h1, g_h1m, W_match, zbuf + 3 * MSV, DV);  // om1
  matchNN(h2f, h2m, zbuf + 4 * MSV);            // miu2g
  matchVec(h2f, g_h2m, zbuf + 5 * MSV);         // phi2g
  matchVec(h2m, g_h2, zbuf + 6 * MSV);          // psi2g
  k_om<<<1, MSV, 0, stream>>>(g_h2, g_h2m, W_match, zbuf + 7 * MSV, DV);  // om2

  // ---- MLP head ----
  k_dense<<<(512 + 127) / 128, 128, 0, stream>>>(zbuf, mw1, mb1, mlp1, 2048, 512, 1);
  k_dense<<<(256 + 127) / 128, 128, 0, stream>>>(mlp1, mw2, mb2, mlp2, 512, 256, 1);
  k_dense<<<1, 128, 0, stream>>>(mlp2, mw3, mb3, mlp3, 256, 128, 1);
  k_dense<<<1, 32, 0, stream>>>(mlp3, mw4, mb4, mlp4, 128, 1, 2);
  k_final<<<1, 32, 0, stream>>>(mlp4, label, (float*)d_out, out_size);
}
